// LSTMModel_2_39814346834406
// MI455X (gfx1250) — matrix operand order repaired and checked by reading, not compiled, not run on hardware
//
#include <hip/hip_runtime.h>
#include <cstddef>

// ---------------------------------------------------------------------------
// Fused 2-layer LSTM + MLP head for MI455X (gfx1250, wave32, WMMA).
//
// B=4096, T=168, P=16, H=24 (4H=96 gates). One wave == one 16-row batch tile,
// runs the whole sequence for both layers + the MLP head. 24 bf16 WMMAs per
// timestep (6 gate-tiles x 2 GEMMs x 2 layers), f32 accumulation.
// Activations use v_exp_f32 + v_rcp_f32 (TRANS ops) to keep the serial
// recurrence chain short -- no IEEE division sequences.
// ---------------------------------------------------------------------------

typedef __attribute__((ext_vector_type(16))) __bf16 v16bf;
typedef __attribute__((ext_vector_type(8)))  float  v8f;

#define NB   4096
#define NT   168
#define NP   16
#define NH   24
#define NG   96     // 4*NH
#define NTL  6      // gate column tiles of 16

__device__ __forceinline__ float fast_rcp(float x) {
  return __builtin_amdgcn_rcpf(x);         // v_rcp_f32 (TRANS)
}
__device__ __forceinline__ float fast_sigmoid(float x) {
  return fast_rcp(1.0f + __expf(-x));      // v_exp_f32 + v_rcp_f32
}
__device__ __forceinline__ float fast_tanh(float x) {
  // tanh(x) = 1 - 2/(e^(2x)+1); stable: e->inf gives 1, e->0 gives -1
  float e = __expf(2.0f * x);
  return 1.0f - 2.0f * fast_rcp(e + 1.0f);
}

// One GEMM phase: g[16x96] = A(16xK)*Bi + h(16x24)*Bh + bias, staged to LDS
// column-major gbuf[col*16 + row] for the pointwise phase.
__device__ __forceinline__ void lstm_gemm_step(
    const v16bf& ain, const v16bf& hstate,
    const v16bf* __restrict__ Bi, const v16bf* __restrict__ Bh,
    const float* __restrict__ biasv, float* __restrict__ gbuf,
    int lx, int half)
{
#pragma unroll
  for (int n = 0; n < NTL; ++n) {
    v8f acc;
#pragma unroll
    for (int e = 0; e < 8; ++e) acc[e] = biasv[n];   // bias broadcast (lane owns one column)
    acc = __builtin_amdgcn_wmma_f32_16x16x32_bf16(false, ain,    false, Bi[n],
                                                  (short)0, acc, false, false);
    acc = __builtin_amdgcn_wmma_f32_16x16x32_bf16(false, hstate, false, Bh[n],
                                                  (short)0, acc, false, false);
    // C layout: lane holds rows rb..rb+7 (rb = half*8) of column c0.
    const int c0 = n * 16 + lx;
    float* p = gbuf + c0 * 16 + half * 8;
    *reinterpret_cast<float4*>(p)     = make_float4(acc[0], acc[1], acc[2], acc[3]);
    *reinterpret_cast<float4*>(p + 4) = make_float4(acc[4], acc[5], acc[6], acc[7]);
  }
}

// Pointwise LSTM cell update. Lane->unit mapping matches the bf16 A-matrix
// layout, so the new h state (and tanh(h) layer output) are produced directly
// in WMMA A-operand register layout:
//   lanes 0-15 : row=lx, j = idx (idx<8) / idx+8 (idx>=8)   -> elems 0..15
//   lanes16-31 : row=lx, j = idx+8 (idx<8), elems 8..15 = 0 (K pad 24..31)
template <bool WANT_OUT>
__device__ __forceinline__ void lstm_elem_step(
    const float* __restrict__ gbuf, float* __restrict__ c,
    v16bf& hstate, v16bf& hout, int lx, int half)
{
#pragma unroll
  for (int idx = 0; idx < 16; ++idx) {
    const int  j      = (half == 0) ? ((idx < 8) ? idx : idx + 8) : (idx + 8);
    const bool active = (half == 0) || (idx < 8);
    const float gi = gbuf[(0 * NH + j) * 16 + lx];
    const float gf = gbuf[(1 * NH + j) * 16 + lx];
    const float gg = gbuf[(2 * NH + j) * 16 + lx];
    const float go = gbuf[(3 * NH + j) * 16 + lx];
    float cn = fast_sigmoid(gf) * c[idx] + fast_sigmoid(gi) * fast_tanh(gg);
    float hv = fast_sigmoid(go) * fast_tanh(cn);
    cn = active ? cn : 0.0f;
    hv = active ? hv : 0.0f;
    c[idx] = cn;
    hstate[idx] = (__bf16)hv;
    if (WANT_OUT) hout[idx] = (__bf16)(active ? fast_tanh(hv) : 0.0f);
  }
}

__global__ __launch_bounds__(32) void lstm2_mlp_fused(
    const float* __restrict__ x,
    const float* __restrict__ W_ih1, const float* __restrict__ W_hh1,
    const float* __restrict__ b_ih1, const float* __restrict__ b_hh1,
    const float* __restrict__ W_ih2, const float* __restrict__ W_hh2,
    const float* __restrict__ b_ih2, const float* __restrict__ b_hh2,
    const float* __restrict__ W1, const float* __restrict__ b1,
    const float* __restrict__ W2, const float* __restrict__ b2,
    float* __restrict__ out)
{
  __shared__ float gbuf[NG * 16];            // 6 KB gate staging (col-major)

  const int lane = threadIdx.x;              // wave32
  const int lx   = lane & 15;                // row (elementwise) / N column (GEMM)
  const int half = lane >> 4;                // A/B-matrix K half selector
  const int base = blockIdx.x * 16;          // batch tile base row

  // ---- Preload all weights into registers in WMMA B-matrix layout ----------
  // B (32x16 bf16): lane n-col = lx + 16n; elem e <-> K = half*8 + e (e<8), 16 + half*8 + (e-8) (e>=8).
  v16bf Bih1[NTL], Bhh1[NTL], Bih2[NTL], Bhh2[NTL];
  float bias1[NTL], bias2[NTL];
#pragma unroll
  for (int n = 0; n < NTL; ++n) {
    const int N = n * 16 + lx;               // gate row of W (0..95)
#pragma unroll
    for (int e = 0; e < 16; ++e) {
      const int K = (e < 8) ? (half * 8 + e) : (16 + half * 8 + (e - 8));
      Bih1[n][e] = (__bf16)((K < NP) ? W_ih1[N * NP + K] : 0.0f);
      Bhh1[n][e] = (__bf16)((K < NH) ? W_hh1[N * NH + K] : 0.0f);
      Bih2[n][e] = (__bf16)((K < NH) ? W_ih2[N * NH + K] : 0.0f);
      Bhh2[n][e] = (__bf16)((K < NH) ? W_hh2[N * NH + K] : 0.0f);
    }
    bias1[n] = b_ih1[N] + b_hh1[N];
    bias2[n] = b_ih2[N] + b_hh2[N];
  }

  // ---- State ---------------------------------------------------------------
  v16bf h1s, h1o, h2s, dummy;
  float c1[16], c2[16];
#pragma unroll
  for (int e = 0; e < 16; ++e) {
    h1s[e] = (__bf16)0.0f; h1o[e] = (__bf16)0.0f;
    h2s[e] = (__bf16)0.0f; dummy[e] = (__bf16)0.0f;
    c1[e] = 0.0f; c2[e] = 0.0f;
  }

  // x A-operand: lanes 0-15 load features 0..7 of row lx, lanes 16-31 load
  // features 8..15 (K octets per ISA A layout); K 16..31 are zero pad.
  const float* xp = x + (size_t)(base + lx) * NT * NP + half * 8;

  // ---- Sequential recurrence over T, both layers fused ---------------------
#pragma unroll 1
  for (int t = 0; t < NT; ++t) {
    const float4 f0 = *reinterpret_cast<const float4*>(xp + t * NP);
    const float4 f1 = *reinterpret_cast<const float4*>(xp + t * NP + 4);
    if (t + 1 < NT)
      __builtin_prefetch(xp + (t + 1) * NP, 0, 0);   // global_prefetch_b8

    v16bf ax;
    ax[0] = (__bf16)f0.x; ax[1] = (__bf16)f0.y; ax[2] = (__bf16)f0.z; ax[3] = (__bf16)f0.w;
    ax[4] = (__bf16)f1.x; ax[5] = (__bf16)f1.y; ax[6] = (__bf16)f1.z; ax[7] = (__bf16)f1.w;
#pragma unroll
    for (int e = 8; e < 16; ++e) ax[e] = (__bf16)0.0f;

    // Layer 1: g = x_t*W_ih1^T + h1*W_hh1^T + b  -> update (c1, h1s), h1o = tanh(h1)
    lstm_gemm_step(ax, h1s, Bih1, Bhh1, bias1, gbuf, lx, half);
    lstm_elem_step<true>(gbuf, c1, h1s, h1o, lx, half);
    // Layer 2: input is tanh(layer-1 h) already in A layout
    lstm_gemm_step(h1o, h2s, Bih2, Bhh2, bias2, gbuf, lx, half);
    lstm_elem_step<false>(gbuf, c2, h2s, dummy, lx, half);
  }

  // ---- MLP head on tanh(h2[T-1]) ------------------------------------------
  // Stage tanh(h2 state) row-major [16][24] into (reused) LDS.
#pragma unroll
  for (int idx = 0; idx < 16; ++idx) {
    const int  j      = (half == 0) ? ((idx < 8) ? idx : idx + 8) : (idx + 8);
    const bool active = (half == 0) || (idx < 8);
    if (active) gbuf[lx * NH + j] = fast_tanh((float)h2s[idx]);
  }
  __syncthreads();   // single-wave block: S_NOP, DS is in-order within a wave

  if (half == 0) {   // lanes 0..15 each finish one batch row
    float hrow[NH];
#pragma unroll
    for (int j = 0; j < NH; ++j) hrow[j] = gbuf[lx * NH + j];
    float m[16];
#pragma unroll
    for (int k = 0; k < 16; ++k) {
      float s = b1[k];
#pragma unroll
      for (int j = 0; j < NH; ++j) s += hrow[j] * W1[k * NH + j];
      m[k] = fmaxf(s, 0.0f);
    }
#pragma unroll
    for (int k2 = 0; k2 < NH; ++k2) {
      float s = b2[k2];
#pragma unroll
      for (int k = 0; k < 16; ++k) s += m[k] * W2[k2 * 16 + k];
      out[(size_t)(base + lx) * NH + k2] = s;
    }
  }
}

extern "C" void kernel_launch(void* const* d_in, const int* in_sizes, int n_in,
                              void* d_out, int out_size, void* d_ws, size_t ws_size,
                              hipStream_t stream) {
  (void)in_sizes; (void)n_in; (void)out_size; (void)d_ws; (void)ws_size;
  const float* x     = (const float*)d_in[0];
  const float* W_ih1 = (const float*)d_in[1];
  const float* W_hh1 = (const float*)d_in[2];
  const float* b_ih1 = (const float*)d_in[3];
  const float* b_hh1 = (const float*)d_in[4];
  const float* W_ih2 = (const float*)d_in[5];
  const float* W_hh2 = (const float*)d_in[6];
  const float* b_ih2 = (const float*)d_in[7];
  const float* b_hh2 = (const float*)d_in[8];
  const float* W1    = (const float*)d_in[9];
  const float* b1    = (const float*)d_in[10];
  const float* W2    = (const float*)d_in[11];
  const float* b2    = (const float*)d_in[12];
  float* out = (float*)d_out;

  // One wave32 block per 16-row batch tile: 4096/16 = 256 independent blocks.
  lstm2_mlp_fused<<<NB / 16, 32, 0, stream>>>(
      x, W_ih1, W_hh1, b_ih1, b_hh1, W_ih2, W_hh2, b_ih2, b_hh2,
      W1, b1, W2, b2, out);
}
